// DPPSearch_11012296147230
// MI455X (gfx1250) — matrix operand order repaired and checked by reading, not compiled, not run on hardware
//
#include <hip/hip_runtime.h>
#include <math.h>

#define BB 16
#define LL 48
#define VV 50257
#define DD 512
#define TOPK 16
#define NIT 8
#define RW 0.1f
#define EPSV 1e-12f

typedef __attribute__((ext_vector_type(16))) __bf16 v16bf;
typedef __attribute__((ext_vector_type(8)))  float  v8f;

union FragU { unsigned u[8]; v16bf v; };

// ---------------------------------------------------------------------------
// threefry2x32 (JAX-style rotation schedule), deterministic counter RNG
// ---------------------------------------------------------------------------
__device__ __forceinline__ void threefry2x32(unsigned k0, unsigned k1,
                                             unsigned c0, unsigned c1,
                                             unsigned& o0, unsigned& o1) {
  const unsigned rotA[4] = {13u, 15u, 26u, 6u};
  const unsigned rotB[4] = {17u, 29u, 16u, 24u};
  unsigned ks[3] = {k0, k1, k0 ^ k1 ^ 0x1BD11BDAu};
  unsigned x0 = c0 + ks[0];
  unsigned x1 = c1 + ks[1];
  #pragma unroll
  for (int g = 0; g < 5; ++g) {
    const unsigned* rot = (g & 1) ? rotB : rotA;
    #pragma unroll
    for (int r = 0; r < 4; ++r) {
      x0 += x1;
      x1 = (x1 << rot[r]) | (x1 >> (32u - rot[r]));
      x1 ^= x0;
    }
    x0 += ks[(g + 1) % 3];
    x1 += ks[(g + 2) % 3] + (unsigned)(g + 1);
  }
  o0 = x0; o1 = x1;
}

// ---------------------------------------------------------------------------
// Kernel 1: per-row top-16 (values+indices), argmax, and row sum of probas.
// One block per (b,l) row; fused so probas is read once here.
// ---------------------------------------------------------------------------
__global__ __launch_bounds__(256)
void topk_rowsum_kernel(const float* __restrict__ probas,
                        const int*   __restrict__ mask,
                        float* __restrict__ topkP, int* __restrict__ topkI,
                        int*   __restrict__ MAPo,  float* __restrict__ rowsum) {
  __shared__ float mv[256 * TOPK];
  __shared__ int   mi[256 * TOPK];
  __shared__ float red[256];
  const int row = blockIdx.x;
  const int tid = threadIdx.x;
  const float* rp = probas + (size_t)row * VV;

  float* my  = &mv[tid * TOPK];
  int*   myI = &mi[tid * TOPK];
  #pragma unroll
  for (int k = 0; k < TOPK; ++k) { my[k] = -3.4e38f; myI[k] = 0; }

  float thr = -3.4e38f;
  float sum = 0.f;
  for (int v = tid; v < VV; v += 256) {
    __builtin_prefetch(rp + v + 256 * 8, 0, 0);   // -> global_prefetch_b8
    const float p = rp[v];
    sum += p;
    if (p > thr) {                                 // rare insertion path
      int pos = TOPK - 1;
      while (pos > 0 && p > my[pos - 1]) {
        my[pos] = my[pos - 1]; myI[pos] = myI[pos - 1]; --pos;
      }
      my[pos] = p; myI[pos] = v;
      thr = my[TOPK - 1];
    }
  }
  red[tid] = sum;
  __syncthreads();
  for (int off = 128; off > 0; off >>= 1) {
    if (tid < off) red[tid] += red[tid + off];
    __syncthreads();
  }
  // pairwise merge of sorted-descending 16-lists: 8 steps -> slot 0 = top-16
  for (int s = 128; s > 0; s >>= 1) {
    if (tid < s) {
      const float* bv = &mv[(tid + s) * TOPK];
      const int*   bI = &mi[(tid + s) * TOPK];
      float rv[TOPK]; int ri[TOPK];
      int i = 0, j = 0;
      #pragma unroll
      for (int k = 0; k < TOPK; ++k) {
        const float va = (i < TOPK) ? my[i] : -3.4e38f;
        const float vb = (j < TOPK) ? bv[j] : -3.4e38f;
        if (va >= vb) { rv[k] = va; ri[k] = (i < TOPK) ? myI[i] : 0; ++i; }
        else          { rv[k] = vb; ri[k] = (j < TOPK) ? bI[j]  : 0; ++j; }
      }
      #pragma unroll
      for (int k = 0; k < TOPK; ++k) { my[k] = rv[k]; myI[k] = ri[k]; }
    }
    __syncthreads();
  }
  if (tid < TOPK) {
    float p = mv[tid];
    if (mask[row] == 0) p = 1.0f;                  // masked rows: uniform top-k
    topkP[row * TOPK + tid] = p;
    topkI[row * TOPK + tid] = mi[tid];
  }
  if (tid == 0) { MAPo[row] = mi[0]; rowsum[row] = red[0]; }
}

// ---------------------------------------------------------------------------
// Kernel 2: one block per (batch, iteration). Gumbel-sample choices, gather
// embeddings to LDS as bf16, Gram via v_wmma_f32_16x16x32_bf16, Cholesky
// logdet in LDS. 128 independent blocks (iterations parallelized).
// ---------------------------------------------------------------------------
__global__ __launch_bounds__(256)
void sample_score_kernel(const float* __restrict__ topkP,
                         const int*   __restrict__ topkI,
                         const int*   __restrict__ MAPo,
                         const int*   __restrict__ mask,
                         const float* __restrict__ embTab,
                         const float* __restrict__ embScale,
                         int*   __restrict__ samples,
                         float* __restrict__ scores) {
  __shared__ unsigned uE[LL * 260];   // 48 rows x 512 bf16, packed pairs, stride 260 dwords
  __shared__ float    sK[LL * LL];    // Gram / Cholesky workspace
  __shared__ int      sSmp[LL];
  __shared__ float    red[64];
  __shared__ int      sLen;

  const int tid = threadIdx.x;
  const int b   = blockIdx.x / NIT;
  const int it  = blockIdx.x % NIT;

  if (tid == 0) {
    int len = 0;
    for (int l = 0; l < LL; ++l) len += (mask[b * LL + l] != 0);
    sLen = len;                       // contiguous-prefix valid length
  }
  if (tid < LL) {
    const int l   = tid;
    const int row = b * LL + l;
    int smp;
    if (l == 0) {
      smp = MAPo[b * LL];             // BOS forced to argmax
    } else {
      unsigned k0, k1;
      threefry2x32(0u, 1u, (unsigned)it, 0x243F6A88u, k0, k1);   // fold_in(it)
      float best = -3.4e38f; int bi = 0;
      unsigned r0 = 0, r1 = 0;
      #pragma unroll
      for (int k = 0; k < TOPK; ++k) {
        if ((k & 1) == 0)
          threefry2x32(k0, k1, (unsigned)(row * 8 + (k >> 1)), 0x85A308D3u, r0, r1);
        const unsigned u = (k & 1) ? r1 : r0;
        const float uf = ((u >> 8) + 0.5f) * (1.0f / 16777216.0f);   // (0,1)
        const float g  = -logf(-logf(uf));                           // Gumbel
        const float t  = logf(topkP[row * TOPK + k]) + g;
        if (t > best) { best = t; bi = k; }
      }
      smp = topkI[row * TOPK + bi];
    }
    sSmp[l] = smp;
    samples[(b * NIT + it) * LL + l] = smp;
  }
  __syncthreads();

  // gather 48 x 512 embedding rows, scale, convert to bf16 pairs in LDS
  const float esc = embScale[0];
  for (int r = 0; r < LL; ++r) {
    const float* ep = embTab + (size_t)sSmp[r] * DD + tid * 2;
    const float x0 = ep[0] * esc;
    const float x1 = ep[1] * esc;
    union { __bf16 h[2]; unsigned u; } pk;
    pk.h[0] = (__bf16)x0; pk.h[1] = (__bf16)x1;
    uE[r * 260 + tid] = pk.u;
  }
  __syncthreads();

  // Gram K = E * E^T : 3x3 tiles of 16x16, one wave per tile (wave 0 takes #8)
  const int wave = tid >> 5, wl = tid & 31;
  const int half = wl >> 4,  lm = wl & 15;
  for (int t = wave; t < 9; t += 8) {
    const int iBase = (t / 3) * 16, jBase = (t % 3) * 16;
    v8f c = {};
    const int arow = iBase + lm, brow = jBase + lm;
    for (int kb = 0; kb < DD / 32; ++kb) {
      const int kBase = kb * 32;
      FragU a, bm;
      #pragma unroll
      for (int v = 0; v < 8; ++v) {
        const int ka  = kBase + 2 * v + 16 * half;                      // A 16x32 layout (contiguous k, matches B)
        const int kbb = kBase + 2 * v + 16 * half;                      // B 32x16 layout
        a.u[v]  = uE[arow * 260 + (ka  >> 1)];
        bm.u[v] = uE[brow * 260 + (kbb >> 1)];
      }
      c = __builtin_amdgcn_wmma_f32_16x16x32_bf16(false, a.v, false, bm.v,
                                                  (short)0, c, false, false);
    }
    #pragma unroll
    for (int r = 0; r < 8; ++r) {
      const int m = iBase + r + 8 * half;          // C/D layout: lanes16-31 -> M+8
      sK[m * LL + jBase + lm] = c[r];
    }
  }
  __syncthreads();

  // identity-pad outside valid len x len block (logdet == logdet of K[:len,:len])
  const int len = sLen;
  for (int idx = tid; idx < LL * LL; idx += 256) {
    const int i = idx / LL, j = idx % LL;
    if (i >= len || j >= len) sK[idx] = (i == j) ? 1.0f : 0.0f;
  }
  __syncthreads();

  // in-LDS Cholesky (lower), right-looking
  for (int k = 0; k < LL; ++k) {
    if (tid == 0) sK[k * LL + k] = sqrtf(sK[k * LL + k]);
    __syncthreads();
    const float dkk = sK[k * LL + k];
    for (int i = k + 1 + tid; i < LL; i += 256) sK[i * LL + k] /= dkk;
    __syncthreads();
    const int n = LL - k - 1;
    for (int idx = tid; idx < n * n; idx += 256) {
      const int i = k + 1 + idx / n;
      const int j = k + 1 + idx % n;
      if (j <= i) sK[i * LL + j] -= sK[i * LL + k] * sK[j * LL + k];
    }
    __syncthreads();
  }
  float lg = 0.f;
  if (tid < LL) lg = logf(sK[tid * LL + tid]);
  if (tid < 64) red[tid] = lg;
  __syncthreads();
  for (int off = 32; off > 0; off >>= 1) {
    if (tid < off) red[tid] += red[tid + off];
    __syncthreads();
  }
  if (tid == 0) scores[b * NIT + it] = 2.0f * red[0];   // logdet = 2*sum(log diag L)
}

// ---------------------------------------------------------------------------
// Kernel 3: argmax over iterations (strict >, earliest iteration wins ties,
// matching the reference's sequential update semantics).
// ---------------------------------------------------------------------------
__global__ void select_best_kernel(const float* __restrict__ scores,
                                   int* __restrict__ bestIt,
                                   float* __restrict__ outScore) {
  const int b = threadIdx.x;
  if (b < BB) {
    float cur = -3.4e38f; int bi = 0;
    for (int it = 0; it < NIT; ++it) {
      const float s = scores[b * NIT + it];
      if (s > cur) { cur = s; bi = it; }
    }
    bestIt[b]   = bi;
    outScore[b] = cur;
  }
}

// ---------------------------------------------------------------------------
// Kernel 4: new_probas = probas * assign / nm with nm from the fused row sum:
// sum(x) = (1-w)*S + (2w-1)*p_best. Streams probas once, writes output once.
// ---------------------------------------------------------------------------
__global__ __launch_bounds__(256)
void redistribute_kernel(const float* __restrict__ probas,
                         const int*   __restrict__ mask,
                         const int*   __restrict__ samples,
                         const int*   __restrict__ bestIt,
                         const float* __restrict__ rowsum,
                         float* __restrict__ out) {
  const int row = blockIdx.x;
  const int b   = row / LL, l = row % LL;
  const int tid = threadIdx.x;
  const int best = samples[(b * NIT + bestIt[b]) * LL + l];
  const float pb = probas[(size_t)row * VV + best];
  float nm = (1.0f - RW) * rowsum[row] + (2.0f * RW - 1.0f) * pb;
  if (mask[row] == 0) nm = EPSV;
  const float inv = 1.0f / nm;
  const float* rp = probas + (size_t)row * VV;
  float*       op = out    + (size_t)row * VV;
  for (int v = tid; v < VV; v += 256) {
    __builtin_prefetch(rp + v + 256 * 8, 0, 0);
    const float p = rp[v];
    const float a = (v == best) ? RW : (1.0f - RW);
    op[v] = p * a * inv;
  }
}

// ---------------------------------------------------------------------------
extern "C" void kernel_launch(void* const* d_in, const int* in_sizes, int n_in,
                              void* d_out, int out_size, void* d_ws, size_t ws_size,
                              hipStream_t stream) {
  (void)in_sizes; (void)n_in; (void)out_size; (void)ws_size;
  const float* probas   = (const float*)d_in[0];
  /* d_in[1] = h_d: unused by the reference */
  const int*   mask     = (const int*)d_in[2];
  const float* embTab   = (const float*)d_in[3];
  const float* embScale = (const float*)d_in[4];
  float* out = (float*)d_out;

  char* ws = (char*)d_ws;
  size_t off = 0;
  auto alloc = [&](size_t bytes) -> char* {
    char* p = ws + off; off += (bytes + 255) & ~(size_t)255; return p;
  };
  float* topkP   = (float*)alloc((size_t)BB * LL * TOPK * sizeof(float));
  int*   topkI   = (int*)  alloc((size_t)BB * LL * TOPK * sizeof(int));
  int*   MAPo    = (int*)  alloc((size_t)BB * LL * sizeof(int));
  float* rowsum  = (float*)alloc((size_t)BB * LL * sizeof(float));
  int*   samples = (int*)  alloc((size_t)BB * NIT * LL * sizeof(int));
  float* scores  = (float*)alloc((size_t)BB * NIT * sizeof(float));
  int*   bestIt  = (int*)  alloc((size_t)BB * sizeof(int));

  topk_rowsum_kernel<<<BB * LL, 256, 0, stream>>>(probas, mask, topkP, topkI,
                                                  MAPo, rowsum);
  sample_score_kernel<<<BB * NIT, 256, 0, stream>>>(topkP, topkI, MAPo, mask,
                                                    embTab, embScale, samples,
                                                    scores);
  select_best_kernel<<<1, 32, 0, stream>>>(scores, bestIt,
                                           out + (size_t)BB * LL * VV);
  redistribute_kernel<<<BB * LL, 256, 0, stream>>>(probas, mask, samples,
                                                   bestIt, rowsum, out);
}
